// MetaDecoder_6536940224713
// MI455X (gfx1250) — compile-verified, matrix operand order checked by reading, not run on hardware
//
#include <hip/hip_runtime.h>

// MI455X (gfx1250) fused MoE decoder:
//   y[t,b,:] = relu(x[b,t,:] @ W1[i[b]]^T + b1) @ W2[i[b]]^T + b2
// bf16 WMMA (v_wmma_f32_16x16x32_bf16), f32 accumulate, fused over DH chunks
// so the 134 MB intermediate h never hits HBM. ~400 MB HBM traffic -> ~17us
// roofline floor at 23.3 TB/s.

typedef __attribute__((ext_vector_type(16))) __bf16 v16bf;
typedef __attribute__((ext_vector_type(8)))  __bf16 v8bf;
typedef __attribute__((ext_vector_type(8)))  float  v8f;
typedef __attribute__((ext_vector_type(4)))  float  v4f;
typedef __attribute__((ext_vector_type(2)))  unsigned int u2;

#define N_EXP 64
#define B_    64
#define T_    256
#define DIN   512
#define DH    2048
#define DOUT  512
#define TM    64          // T rows per workgroup
#define KC    32          // DH chunk width
#define NCHUNK (DH / KC)  // 64

union ABFrag {
    v16bf v;
    struct { v8bf lo, hi; } h;
};

// round-to-nearest-even f32 -> bf16, packed pair
__device__ __forceinline__ unsigned pk_bf16(float a, float b) {
    unsigned ua = __float_as_uint(a), ub = __float_as_uint(b);
    ua = (ua + 0x7FFFu + ((ua >> 16) & 1u)) >> 16;
    ub = (ub + 0x7FFFu + ((ub >> 16) & 1u)) >> 16;
    return (ua & 0xFFFFu) | (ub << 16);
}

__device__ __forceinline__ unsigned short cvt_bf16(float a) {
    unsigned ua = __float_as_uint(a);
    return (unsigned short)((ua + 0x7FFFu + ((ua >> 16) & 1u)) >> 16);
}

__device__ __forceinline__ void st_pack4(__bf16* dst, v4f v) {
    u2 t;
    t[0] = pk_bf16(v[0], v[1]);
    t[1] = pk_bf16(v[2], v[3]);
    *(u2*)dst = t;   // ds_store_b64
}

__global__ __launch_bounds__(256)
void moe_fused_kernel(const int* __restrict__ ig, const float* __restrict__ xg,
                      const float* __restrict__ w1g, const float* __restrict__ b1g,
                      const float* __restrict__ w2g, const float* __restrict__ b2g,
                      float* __restrict__ outg)
{
    __shared__ __align__(16) __bf16 x_lds[TM * DIN];    // 64 KB, resident all chunks
    __shared__ __align__(16) __bf16 w1_lds[KC * DIN];   // 32 KB, per-chunk
    __shared__ __align__(16) __bf16 w2_lds[DOUT * KC];  // 32 KB, per-chunk
    __shared__ __align__(16) __bf16 h_lds[TM * KC];     //  4 KB, per-chunk

    const int tid   = threadIdx.x;
    const int wid   = tid >> 5;            // wave id 0..7 (wave32)
    const int lane  = tid & 31;
    const int l16   = lane & 15;
    const int khalf = (lane >> 4) << 3;    // 0 or 8: per-lane K-run split (ISA 7.12.2)

    const int ttile = blockIdx.x;          // 0..3  (T tile)
    const int b     = blockIdx.y;          // 0..63 (sample)
    const int e     = ig[b];               // routed expert

    // GEMM2 wave tiling: 2(M=32) x 4(N=128) over the 64x512 y tile
    const int row_w = wid >> 2;
    const int col_w = wid & 3;
    // GEMM1 wave tiling: 4(M=16) x 2(N=16) over the 64x32 h chunk
    const int m1 = wid & 3;
    const int n1 = wid >> 2;

    // ---- stage x tile (f32 -> bf16) into LDS, once ----
    {
        const float* src = xg + ((size_t)b * T_ + (size_t)ttile * TM) * DIN;
        const v4f* s4 = (const v4f*)src;
        #pragma unroll 4
        for (int i = tid; i < (TM * DIN) / 4; i += 256)
            st_pack4(&x_lds[i * 4], s4[i]);
    }

    const v8f vzero = {0.f, 0.f, 0.f, 0.f, 0.f, 0.f, 0.f, 0.f};
    v8f yacc[2][8];
    #pragma unroll
    for (int mt = 0; mt < 2; ++mt)
        #pragma unroll
        for (int nt = 0; nt < 8; ++nt)
            yacc[mt][nt] = vzero;

    const float* w1base = w1g + (size_t)e * DH * DIN;
    const float* w2base = w2g + (size_t)e * DOUT * DH;

    for (int ch = 0; ch < NCHUNK; ++ch) {
        const int c0 = ch * KC;

        // ---- stage W1 chunk: rows [c0,c0+32) x 512, contiguous in memory ----
        {
            const float* src = w1base + (size_t)c0 * DIN;
            const v4f* s4 = (const v4f*)src;
            #pragma unroll 4
            for (int i = tid; i < (KC * DIN) / 4; i += 256)
                st_pack4(&w1_lds[i * 4], s4[i]);
            if (ch + 1 < NCHUNK)  // global_prefetch_b8 of next chunk
                __builtin_prefetch((const void*)(src + KC * DIN + tid * 64), 0, 0);
        }
        // ---- stage W2 chunk: 512 rows x cols [c0,c0+32), row stride DH ----
        {
            const float* src = w2base + c0;
            #pragma unroll 4
            for (int i = tid; i < (DOUT * KC) / 4; i += 256) {
                const int row = i >> 3;           // KC/4 = 8 float4 per row
                const int c4  = (i & 7) << 2;
                v4f v = *(const v4f*)(src + (size_t)row * DH + c4);
                st_pack4(&w2_lds[row * KC + c4], v);
            }
            if (ch + 1 < NCHUNK)
                __builtin_prefetch((const void*)(src + KC + (size_t)(tid * 2) * DH), 0, 0);
        }
        __syncthreads();

        // ---- GEMM1: h_chunk(64x32) = relu(x(64x512) @ W1chunk^T + b1) ----
        v8f hacc = vzero;
        const int arow1 = m1 * 16 + l16;
        const int brow1 = n1 * 16 + l16;
        #pragma unroll
        for (int k = 0; k < DIN; k += 32) {
            ABFrag a, bb;
            a.h.lo  = *(const v8bf*)&x_lds[arow1 * DIN + k + khalf];
            a.h.hi  = *(const v8bf*)&x_lds[arow1 * DIN + k + khalf + 16];
            bb.h.lo = *(const v8bf*)&w1_lds[brow1 * DIN + k + khalf];
            bb.h.hi = *(const v8bf*)&w1_lds[brow1 * DIN + k + khalf + 16];
            hacc = __builtin_amdgcn_wmma_f32_16x16x32_bf16(
                       false, a.v, false, bb.v, (short)0, hacc, false, false);
        }
        // bias + relu, write h_chunk to LDS as bf16 (C/D layout: lane=N, vgpr=M)
        {
            const float bias = b1g[(size_t)e * DH + c0 + n1 * 16 + l16];
            const int hn = n1 * 16 + l16;
            #pragma unroll
            for (int r = 0; r < 8; ++r) {
                float hv = hacc[r] + bias;
                hv = hv > 0.f ? hv : 0.f;
                const int hm = m1 * 16 + r + ((lane >> 4) << 3);
                ((unsigned short*)h_lds)[hm * KC + hn] = cvt_bf16(hv);
            }
        }
        __syncthreads();

        // ---- GEMM2: yacc(64x512) += h_chunk(64x32) @ W2chunk^T, K=32 ----
        {
            ABFrag a0, a1;
            const int am0 = row_w * 32 + l16;
            const int am1 = am0 + 16;
            a0.h.lo = *(const v8bf*)&h_lds[am0 * KC + khalf];
            a0.h.hi = *(const v8bf*)&h_lds[am0 * KC + khalf + 16];
            a1.h.lo = *(const v8bf*)&h_lds[am1 * KC + khalf];
            a1.h.hi = *(const v8bf*)&h_lds[am1 * KC + khalf + 16];
            #pragma unroll
            for (int nt = 0; nt < 8; ++nt) {
                ABFrag bb;
                const int bn = col_w * 128 + nt * 16 + l16;
                bb.h.lo = *(const v8bf*)&w2_lds[bn * KC + khalf];
                bb.h.hi = *(const v8bf*)&w2_lds[bn * KC + khalf + 16];
                yacc[0][nt] = __builtin_amdgcn_wmma_f32_16x16x32_bf16(
                                  false, a0.v, false, bb.v, (short)0, yacc[0][nt], false, false);
                yacc[1][nt] = __builtin_amdgcn_wmma_f32_16x16x32_bf16(
                                  false, a1.v, false, bb.v, (short)0, yacc[1][nt], false, false);
            }
        }
        __syncthreads();
    }

    // ---- epilogue: + b2, store y to out[T, B, DOUT] (coalesced along DOUT) ----
    #pragma unroll
    for (int mt = 0; mt < 2; ++mt) {
        #pragma unroll
        for (int nt = 0; nt < 8; ++nt) {
            const int o = col_w * 128 + nt * 16 + l16;
            const float bias2 = b2g[(size_t)e * DOUT + o];
            #pragma unroll
            for (int r = 0; r < 8; ++r) {
                const int t = ttile * TM + row_w * 32 + mt * 16 + r + ((lane >> 4) << 3);
                outg[((size_t)t * B_ + b) * DOUT + o] = yacc[mt][nt][r] + bias2;
            }
        }
    }
}

extern "C" void kernel_launch(void* const* d_in, const int* in_sizes, int n_in,
                              void* d_out, int out_size, void* d_ws, size_t ws_size,
                              hipStream_t stream) {
    const int*   ig  = (const int*)d_in[0];
    const float* xg  = (const float*)d_in[1];
    const float* w1g = (const float*)d_in[2];
    const float* b1g = (const float*)d_in[3];
    const float* w2g = (const float*)d_in[4];
    const float* b2g = (const float*)d_in[5];
    float* outg = (float*)d_out;
    (void)in_sizes; (void)n_in; (void)out_size; (void)d_ws; (void)ws_size;

    dim3 grid(T_ / TM, B_);   // 4 x 64 workgroups
    dim3 block(256);          // 8 wave32 waves
    hipLaunchKernelGGL(moe_fused_kernel, grid, block, 0, stream,
                       ig, xg, w1g, b1g, w2g, b2g, outg);
}
